// CausalReasoningModule_18330920419659
// MI455X (gfx1250) — matrix operand order repaired and checked by reading, not compiled, not run on hardware
//
#include <hip/hip_runtime.h>
#include <hip/hip_bf16.h>
#include <math.h>

// ---------------------------------------------------------------------------
// CausalReasoningModule forward for MI455X (gfx1250, wave32, WMMA).
// All dense GEMMs run on v_wmma_f32_16x16x32_f16 (f16 in, f32 accum).
// Weights are pre-packed into the WMMA B-fragment lane layout so the GEMM
// inner loop is pure b128 loads + wmma. [B,N,N,H] tensors never materialized.
// ---------------------------------------------------------------------------

typedef _Float16 h16_t;
typedef __attribute__((ext_vector_type(8)))  _Float16 v8h;
typedef __attribute__((ext_vector_type(16))) _Float16 v16h;
typedef __attribute__((ext_vector_type(8)))  float    v8f;

static constexpr int NB = 4096;   // batch
static constexpr int NF = 512;    // obs feat
static constexpr int NA = 64;     // act feat
static constexpr int NN = 8;      // nodes
static constexpr int NH = 128;    // hidden
static constexpr int NL = 3;      // gnn layers

__device__ __forceinline__ float gelu_exact(float x) {
  return 0.5f * x * (1.0f + erff(x * 0.70710678118654752f));
}

// ------------------------- WMMA fragment loaders ---------------------------
// A: 16x32 f16 (MxK), row-major source, lda (elements) multiple of 8.
// lane 0-15: M=lane, K={0..15}+k0 ; lane 16-31: M=lane-16, K={16..31}+k0
// -> two aligned 16B loads, no packing moves.
__device__ __forceinline__ v16h frag_a(const h16_t* __restrict__ A, int lda,
                                       int row0, int k0, int lane) {
  const int m  = row0 + (lane & 15);
  const int kb = k0 + ((lane >> 4) << 4);
  const h16_t* p = A + (size_t)m * lda + kb;
  const v8h lo = *(const v8h*)(p);
  const v8h hi = *(const v8h*)(p + 8);
  return __builtin_shufflevector(lo, hi, 0, 1, 2, 3, 4, 5, 6, 7,
                                 8, 9, 10, 11, 12, 13, 14, 15);
}

// Packed B layout: tile (tk,tn) of a [K,N] matrix occupies 512 halves:
//   dst[tile*512 + lane*16 + j] = W[(tk*32 + (lane>>4)*16 + j) * N + tn*16 + (lane&15)]
// Fragment load = one 32B-aligned v16h load (two global_load_b128).
__global__ void pack_w(const float* __restrict__ src, h16_t* __restrict__ dst,
                       int K, int N) {
  const size_t idx = (size_t)blockIdx.x * 256 + threadIdx.x;
  if (idx >= (size_t)K * N) return;
  const int k = (int)(idx / N), n = (int)(idx % N);
  const int tk = k >> 5, kr = k & 31, tn = n >> 4, nr = n & 15;
  const int lane = nr | ((kr >> 4) << 4);
  const int j = kr & 15;
  dst[(((size_t)tk * (N >> 4)) + tn) * 512 + lane * 16 + j] = (h16_t)src[idx];
}

// ------------------------- generic fused WMMA GEMM -------------------------
// C[M,N] = act( LN( A[M,K] @ W[K,N] + bias ) )  (W pre-packed; LN/act optional)
// One wave computes a (16*MT) x (16*NT) tile. When LN==true, 16*NT must equal
// N (wave owns complete rows -> row LayerNorm via half-wave shuffles).
template <int MT, int NT, bool LN, bool ACT_GELU>
__global__ __launch_bounds__(128) void gemm_f16(
    const h16_t* __restrict__ A, int lda,
    const h16_t* __restrict__ Bp, int N,
    const float* __restrict__ bias,
    const float* __restrict__ gamma, const float* __restrict__ beta,
    float* __restrict__ C32, h16_t* __restrict__ C16, int ldc,
    int M, int K) {
  const int lane = threadIdx.x & 31;
  const int wave = threadIdx.x >> 5;
  const int ntn  = N / (16 * NT);
  const int tiles = (M / (16 * MT)) * ntn;
  const int tid = blockIdx.x * 4 + wave;
  if (tid >= tiles) return;
  const int row0 = (tid / ntn) * (16 * MT);
  const int tn0  = (tid % ntn) * NT;        // column tile index
  const int ntt  = N >> 4;                  // packed tiles per k-row

  v8f acc[MT][NT];
  const v8f vzero = {0.f, 0.f, 0.f, 0.f, 0.f, 0.f, 0.f, 0.f};
#pragma unroll
  for (int mt = 0; mt < MT; ++mt)
#pragma unroll
    for (int t = 0; t < NT; ++t) acc[mt][t] = vzero;

  const int ktiles = K >> 5;
  for (int tk = 0; tk < ktiles; ++tk) {
    v16h a[MT];
#pragma unroll
    for (int mt = 0; mt < MT; ++mt)
      a[mt] = frag_a(A, lda, row0 + 16 * mt, tk << 5, lane);
    const h16_t* bp = Bp + ((size_t)tk * ntt + tn0) * 512 + lane * 16;
#pragma unroll
    for (int t = 0; t < NT; ++t) {
      const v16h b = *(const v16h*)(bp + (size_t)t * 512);
#pragma unroll
      for (int mt = 0; mt < MT; ++mt)
        acc[mt][t] = __builtin_amdgcn_wmma_f32_16x16x32_f16(
            false, a[mt], false, b, (short)0, acc[mt][t], false, false);
    }
  }

  const int nloc = lane & 15;
  float bia[NT], gam[NT], bet[NT];
#pragma unroll
  for (int t = 0; t < NT; ++t) {
    const int n = tn0 * 16 + 16 * t + nloc;
    bia[t] = bias ? bias[n] : 0.f;
    if (LN) { gam[t] = gamma[n]; bet[t] = beta[n]; }
  }

#pragma unroll
  for (int mt = 0; mt < MT; ++mt) {
#pragma unroll
    for (int t = 0; t < NT; ++t)
#pragma unroll
      for (int g = 0; g < 8; ++g) acc[mt][t][g] += bia[t];

    if (LN) {
      // accumulator VGPR g holds output row g + 8*(lane>=16) of this m-tile;
      // its 16*NT columns live across this half-wave's 16 lanes and NT frags.
      float mean[8], rstd[8];
#pragma unroll
      for (int g = 0; g < 8; ++g) {
        float s = 0.f, q = 0.f;
#pragma unroll
        for (int t = 0; t < NT; ++t) { const float x = acc[mt][t][g]; s += x; q += x * x; }
#pragma unroll
        for (int m = 1; m < 16; m <<= 1) {
          s += __shfl_xor(s, m, 32);
          q += __shfl_xor(q, m, 32);
        }
        const float inv = 1.f / (16.f * NT);
        const float mu = s * inv;
        mean[g] = mu;
        rstd[g] = rsqrtf(q * inv - mu * mu + 1e-5f);
      }
#pragma unroll
      for (int t = 0; t < NT; ++t)
#pragma unroll
        for (int g = 0; g < 8; ++g)
          acc[mt][t][g] = (acc[mt][t][g] - mean[g]) * rstd[g] * gam[t] + bet[t];
    }
    if (ACT_GELU) {
#pragma unroll
      for (int t = 0; t < NT; ++t)
#pragma unroll
        for (int g = 0; g < 8; ++g) acc[mt][t][g] = gelu_exact(acc[mt][t][g]);
    }

    const int mb = row0 + 16 * mt + ((lane >> 4) << 3);
#pragma unroll
    for (int t = 0; t < NT; ++t) {
      const int n = tn0 * 16 + 16 * t + nloc;
#pragma unroll
      for (int g = 0; g < 8; ++g) {
        const size_t idx = (size_t)(mb + g) * ldc + n;
        if (C32) C32[idx] = acc[mt][t][g];
        if (C16) C16[idx] = (h16_t)acc[mt][t][g];
      }
    }
  }
}

// ------------------------------ prep kernels -------------------------------
__global__ void build_x(const float* __restrict__ obs, const float* __restrict__ act,
                        const float* __restrict__ noise, h16_t* __restrict__ xcat) {
  const size_t idx = (size_t)blockIdx.x * 256 + threadIdx.x;
  const size_t total = (size_t)2 * NB * (NF + NA);
  if (idx >= total) return;
  const size_t r = idx / (NF + NA);
  const int c = (int)(idx % (NF + NA));
  const size_t rb = (r < NB) ? r : r - NB;
  float v;
  if (c < NF) v = obs[rb * NF + c];
  else {
    v = act[rb * NA + (c - NF)];
    if (r >= NB) v += 0.1f * noise[rb * NA + (c - NF)];
  }
  xcat[idx] = (h16_t)v;
}

__global__ void cvt16(const float* __restrict__ src, h16_t* __restrict__ dst, int n) {
  const int i = blockIdx.x * 256 + threadIdx.x;
  if (i < n) dst[i] = (h16_t)src[i];
}

// --------------------- pairwise score -> adjacency (fused) -----------------
__global__ __launch_bounds__(256) void pair_adj(
    const float* __restrict__ hi, const float* __restrict__ hj,
    const float* __restrict__ es1_b, const float* __restrict__ es2_w,
    const float* __restrict__ es2_b, const float* __restrict__ temp,
    float* __restrict__ adj) {
  const int b = blockIdx.x;
  const int wave = threadIdx.x >> 5, lane = threadIdx.x & 31;
  const float tp = fabsf(temp[0]) + 1e-6f;
  for (int p = wave; p < NN * NN; p += 8) {
    const int i = p >> 3, j = p & 7;
    const float* pi = hi + ((size_t)b * NN + i) * NH;
    const float* pj = hj + ((size_t)b * NN + j) * NH;
    float s = 0.f;
#pragma unroll
    for (int q = 0; q < 4; ++q) {
      const int n = lane * 4 + q;
      s += gelu_exact(pi[n] + pj[n] + es1_b[n]) * es2_w[n];
    }
#pragma unroll
    for (int m = 1; m < 32; m <<= 1) s += __shfl_xor(s, m, 32);
    if (lane == 0) {
      const float sc = (i == j) ? -1e9f : (s + es2_b[0]);
      adj[(size_t)b * 64 + p] = 1.f / (1.f + expf(-sc / tp));
    }
  }
}

// --------------- GNN message + LN + GELU + adj-weighted agg ----------------
__global__ __launch_bounds__(256) void msg_agg(
    const float* __restrict__ hjg, const float* __restrict__ hig,
    const float* __restrict__ gmb, const float* __restrict__ gmg,
    const float* __restrict__ gmbe, const float* __restrict__ adj,
    const h16_t* __restrict__ vcur, h16_t* __restrict__ acat) {
  const int b = blockIdx.x;
  const int i = threadIdx.x >> 5, lane = threadIdx.x & 31;
  const float* hp = hig + ((size_t)b * NN + i) * NH;
  float accq[4] = {0.f, 0.f, 0.f, 0.f};
  for (int j = 0; j < NN; ++j) {
    const float* jp = hjg + ((size_t)b * NN + j) * NH;
    float x[4], s = 0.f, q = 0.f;
#pragma unroll
    for (int t = 0; t < 4; ++t) {
      const int n = lane * 4 + t;
      x[t] = jp[n] + hp[n] + gmb[n];
      s += x[t]; q += x[t] * x[t];
    }
#pragma unroll
    for (int m = 1; m < 32; m <<= 1) {
      s += __shfl_xor(s, m, 32);
      q += __shfl_xor(q, m, 32);
    }
    const float mu = s * (1.f / NH);
    const float rs = rsqrtf(q * (1.f / NH) - mu * mu + 1e-5f);
    const float a = adj[(size_t)b * 64 + i * 8 + j];
#pragma unroll
    for (int t = 0; t < 4; ++t) {
      const int n = lane * 4 + t;
      accq[t] += a * gelu_exact((x[t] - mu) * rs * gmg[n] + gmbe[n]);
    }
  }
  const size_t r = (size_t)b * NN + i;
#pragma unroll
  for (int t = 0; t < 4; ++t) {
    const int n = lane * 4 + t;
    acat[r * 256 + n]      = vcur[r * NH + n];
    acat[r * 256 + NH + n] = (h16_t)accq[t];
  }
}

// -------------------- pairwise effect estimator (fused) --------------------
__global__ __launch_bounds__(256) void pair_eff(
    const float* __restrict__ hi, const float* __restrict__ hj,
    const float* __restrict__ ee1_b, const float* __restrict__ eg,
    const float* __restrict__ ebe, const float* __restrict__ ee2_w,
    const float* __restrict__ ee2_b, float* __restrict__ eff) {
  const int b = blockIdx.x;
  const int wave = threadIdx.x >> 5, lane = threadIdx.x & 31;
  for (int p = wave; p < NN * NN; p += 8) {
    const int i = p >> 3, j = p & 7;
    const float* pi = hi + ((size_t)b * NN + i) * NH;
    const float* pj = hj + ((size_t)b * NN + j) * NH;
    float x[4], s = 0.f, q = 0.f;
#pragma unroll
    for (int t = 0; t < 4; ++t) {
      const int n = lane * 4 + t;
      x[t] = pi[n] + pj[n] + ee1_b[n];
      s += x[t]; q += x[t] * x[t];
    }
#pragma unroll
    for (int m = 1; m < 32; m <<= 1) {
      s += __shfl_xor(s, m, 32);
      q += __shfl_xor(q, m, 32);
    }
    const float mu = s * (1.f / NH);
    const float rs = rsqrtf(q * (1.f / NH) - mu * mu + 1e-5f);
    float d = 0.f;
#pragma unroll
    for (int t = 0; t < 4; ++t) {
      const int n = lane * 4 + t;
      d += gelu_exact((x[t] - mu) * rs * eg[n] + ebe[n]) * ee2_w[n];
    }
#pragma unroll
    for (int m = 1; m < 32; m <<= 1) d += __shfl_xor(d, m, 32);
    if (lane == 0)
      eff[(size_t)b * 64 + p] = (i == j) ? 0.f : (d + ee2_b[0]);
  }
}

// ------------------------- intervention kernels ----------------------------
__global__ void build_z(const h16_t* __restrict__ vfin16, const h16_t* __restrict__ zv16,
                        h16_t* __restrict__ z16) {
  const size_t idx = (size_t)blockIdx.x * 256 + threadIdx.x;
  if (idx >= (size_t)NB * NN * NH) return;
  const size_t row = idx >> 7;
  const int n = (int)(idx & 127);
  const int i = (int)(row & 7);
  const size_t b = row >> 3;
  z16[idx] = (i == 0) ? zv16[(b << 7) + n] : vfin16[idx];
}

__global__ __launch_bounds__(256) void iv_pre(
    const float* __restrict__ hs, const float* __restrict__ hz,
    const float* __restrict__ b1, const float* __restrict__ g,
    const float* __restrict__ be, h16_t* __restrict__ t16) {
  const size_t row = (size_t)blockIdx.x * 8 + (threadIdx.x >> 5);
  const int lane = threadIdx.x & 31;
  const size_t b = row >> 3;
  float x[4], s = 0.f, q = 0.f;
#pragma unroll
  for (int t = 0; t < 4; ++t) {
    const int n = lane * 4 + t;
    x[t] = hs[(b << 7) + n] + hz[(row << 7) + n] + b1[n];
    s += x[t]; q += x[t] * x[t];
  }
#pragma unroll
  for (int m = 1; m < 32; m <<= 1) {
    s += __shfl_xor(s, m, 32);
    q += __shfl_xor(q, m, 32);
  }
  const float mu = s * (1.f / NH);
  const float rs = rsqrtf(q * (1.f / NH) - mu * mu + 1e-5f);
#pragma unroll
  for (int t = 0; t < 4; ++t) {
    const int n = lane * 4 + t;
    t16[(row << 7) + n] = (h16_t)gelu_exact((x[t] - mu) * rs * g[n] + be[n]);
  }
}

__global__ void iv_post(const float* __restrict__ d, const float* __restrict__ adj,
                        const float* __restrict__ vfin32, const float* __restrict__ zval,
                        float* __restrict__ out) {
  const size_t idx = (size_t)blockIdx.x * 256 + threadIdx.x;
  if (idx >= (size_t)NB * NN * NH) return;
  const size_t b = idx >> 10;
  const int i = (int)((idx >> 7) & 7);
  const int n = (int)(idx & 127);
  const float z = (i == 0) ? zval[(b << 7) + n] : vfin32[idx];
  out[idx] = z + d[idx] * adj[(b << 6) + i];   // adj[b,0,i]
}

// ---------------------- counterfactual input builder -----------------------
__global__ void build_cfin(const h16_t* __restrict__ vfin16, const h16_t* __restrict__ vcf16,
                           const float* __restrict__ vfin32, const float* __restrict__ vcf32,
                           h16_t* __restrict__ cfin) {
  const size_t idx = (size_t)blockIdx.x * 256 + threadIdx.x;
  if (idx >= (size_t)NB * 384) return;
  const size_t b = idx / 384;
  const int c = (int)(idx % 384);
  float val;
  if (c < 128) {
    val = (float)vfin16[(b << 10) + c];                 // v[:,0]
  } else if (c < 256) {
    val = (float)vcf16[(b << 10) + (c - 128)];          // v_cf[:,0]
  } else {
    const int n = c - 256;
    float s = 0.f;
#pragma unroll
    for (int i = 0; i < NN; ++i)
      s += vfin32[((b << 3) + i) * NH + n] - vcf32[((b << 3) + i) * NH + n];
    val = s * 0.125f;                                    // ctx
  }
  cfin[idx] = (h16_t)val;
}

// ----------------------------- adj mean reduce -----------------------------
__global__ __launch_bounds__(256) void reduce_partial(const float* __restrict__ adj,
                                                      float* __restrict__ part) {
  float s = 0.f;
  for (int idx = blockIdx.x * 256 + threadIdx.x; idx < NB * 64; idx += 256 * 256)
    s += adj[idx];
#pragma unroll
  for (int m = 1; m < 32; m <<= 1) s += __shfl_xor(s, m, 32);
  __shared__ float sm[8];
  if ((threadIdx.x & 31) == 0) sm[threadIdx.x >> 5] = s;
  __syncthreads();
  if (threadIdx.x == 0) {
    float t = 0.f;
    for (int w = 0; w < 8; ++w) t += sm[w];
    part[blockIdx.x] = t;
  }
}

__global__ __launch_bounds__(256) void reduce_final(const float* __restrict__ part,
                                                    float* __restrict__ out) {
  float s = part[threadIdx.x];
#pragma unroll
  for (int m = 1; m < 32; m <<= 1) s += __shfl_xor(s, m, 32);
  __shared__ float sm[8];
  if ((threadIdx.x & 31) == 0) sm[threadIdx.x >> 5] = s;
  __syncthreads();
  if (threadIdx.x == 0) {
    float t = 0.f;
    for (int w = 0; w < 8; ++w) t += sm[w];
    out[0] = t / (float)(NB * 64);
  }
}

// ------------------------------- launcher ----------------------------------
extern "C" void kernel_launch(void* const* d_in, const int* in_sizes, int n_in,
                              void* d_out, int out_size, void* d_ws, size_t ws_size,
                              hipStream_t stream) {
  (void)in_sizes; (void)n_in; (void)out_size; (void)ws_size;
  const float* obs    = (const float*)d_in[0];
  const float* act    = (const float*)d_in[1];
  const float* z_val  = (const float*)d_in[2];
  const float* noise  = (const float*)d_in[3];
  const float* ip_w   = (const float*)d_in[4];
  const float* ip_b   = (const float*)d_in[5];
  const float* ve1_w  = (const float*)d_in[6];
  const float* ve1_b  = (const float*)d_in[7];
  const float* ve_g   = (const float*)d_in[8];
  const float* ve_be  = (const float*)d_in[9];
  const float* ve2_w  = (const float*)d_in[10];
  const float* ve2_b  = (const float*)d_in[11];
  const float* es1_w  = (const float*)d_in[12];
  const float* es1_b  = (const float*)d_in[13];
  const float* es2_w  = (const float*)d_in[14];
  const float* es2_b  = (const float*)d_in[15];
  const float* temp   = (const float*)d_in[16];
  const float* gm_w   = (const float*)d_in[17];
  const float* gm_b   = (const float*)d_in[18];
  const float* gm_g   = (const float*)d_in[19];
  const float* gm_be  = (const float*)d_in[20];
  const float* gu_w   = (const float*)d_in[21];
  const float* gu_b   = (const float*)d_in[22];
  const float* gu_g   = (const float*)d_in[23];
  const float* gu_be  = (const float*)d_in[24];
  const float* iv1_w  = (const float*)d_in[25];
  const float* iv1_b  = (const float*)d_in[26];
  const float* iv_g   = (const float*)d_in[27];
  const float* iv_be  = (const float*)d_in[28];
  const float* iv2_w  = (const float*)d_in[29];
  const float* iv2_b  = (const float*)d_in[30];
  const float* cf1_w  = (const float*)d_in[31];
  const float* cf1_b  = (const float*)d_in[32];
  const float* cf_g   = (const float*)d_in[33];
  const float* cf_be  = (const float*)d_in[34];
  const float* cf2_w  = (const float*)d_in[35];
  const float* cf2_b  = (const float*)d_in[36];
  const float* ee1_w  = (const float*)d_in[37];
  const float* ee1_b  = (const float*)d_in[38];
  const float* ee_g   = (const float*)d_in[39];
  const float* ee_be  = (const float*)d_in[40];
  const float* ee2_w  = (const float*)d_in[41];
  const float* ee2_b  = (const float*)d_in[42];
  const float* op_w   = (const float*)d_in[43];
  const float* op_b   = (const float*)d_in[44];

  // ---- output layout (floats) ----
  float* out = (float*)d_out;
  const size_t OFF_ADJ = 0;
  const size_t OFF_V   = OFF_ADJ + (size_t)NB * 64;
  const size_t OFF_EFF = OFF_V + (size_t)NB * NN * NH;
  const size_t OFF_ITV = OFF_EFF + (size_t)NB * 64;
  const size_t OFF_CF  = OFF_ITV + (size_t)NB * NN * NH;
  const size_t OFF_CR  = OFF_CF + (size_t)NB * NH;
  const size_t OFF_MEAN = OFF_CR + (size_t)NB * 64;

  // ---- workspace layout ----
  char* ws = (char*)d_ws;
  size_t off = 0;
  auto carve = [&](size_t bytes) {
    void* p = ws + off;
    off += (bytes + 255) & ~(size_t)255;
    return p;
  };
  // packed f16 weight pool (offsets in halves; packing preserves element count)
  h16_t* W16 = (h16_t*)carve((size_t)933888 * 2);
  const size_t O_IP = 0, O_VE1 = 294912, O_VE2 = 360448,
               O_ES1A = 491520, O_ES1B = 507904,
               O_GM = 524288, O_GU = 622592,
               O_IV1A = 720896, O_IV1B = 737280, O_IV2 = 753664,
               O_CF1 = 770048, O_CF2 = 819200,
               O_EE1A = 835584, O_EE1B = 851968, O_OP = 868352;
  h16_t* zv16   = (h16_t*)carve((size_t)NB * NH * 2);
  h16_t* xcat16 = (h16_t*)carve((size_t)2 * NB * (NF + NA) * 2);  // reused as z16full
  h16_t* x16    = (h16_t*)carve((size_t)2 * NB * NF * 2);
  h16_t* hbuf16 = (h16_t*)carve((size_t)2 * NB * NH * 2);
  float* vb32   = (float*)carve((size_t)2 * NB * NN * NH * 4);
  h16_t* v16    = (h16_t*)carve((size_t)2 * NB * NN * NH * 2);
  float* p1     = (float*)carve((size_t)NB * NN * NH * 4);
  float* p2     = (float*)carve((size_t)NB * NN * NH * 4);
  h16_t* acat16 = (h16_t*)carve((size_t)NB * NN * 2 * NH * 2);    // reused as t16
  h16_t* w16a   = (h16_t*)carve((size_t)NB * NN * NH * 2);
  h16_t* w16b   = (h16_t*)carve((size_t)NB * NN * NH * 2);
  float* hs32   = (float*)carve((size_t)NB * NH * 4);
  h16_t* cfin16 = (h16_t*)carve((size_t)NB * 384 * 2);
  h16_t* c116   = (h16_t*)carve((size_t)NB * NH * 2);
  float* part   = (float*)carve(256 * 4);

  auto pack = [&](const float* src, h16_t* dst, int K, int N) {
    pack_w<<<(K * N + 255) / 256, 256, 0, stream>>>(src, dst, K, N);
  };

  // ---- 0) input staging + weight packing into WMMA-B layout ----
  {
    const size_t total = (size_t)2 * NB * (NF + NA);
    build_x<<<(int)((total + 255) / 256), 256, 0, stream>>>(obs, act, noise, xcat16);
  }
  pack(ip_w,  W16 + O_IP,  NF + NA, NF);
  pack(ve1_w, W16 + O_VE1, NF, NH);
  pack(ve2_w, W16 + O_VE2, NH, NN * NH);
  pack(es1_w,           W16 + O_ES1A, NH, NH);
  pack(es1_w + NH * NH, W16 + O_ES1B, NH, NH);
  for (int l = 0; l < NL; ++l) {
    pack(gm_w + (size_t)l * 2 * NH * NH,           W16 + O_GM + l * 32768, NH, NH);
    pack(gm_w + (size_t)l * 2 * NH * NH + NH * NH, W16 + O_GM + l * 32768 + 16384, NH, NH);
    pack(gu_w + (size_t)l * 2 * NH * NH,           W16 + O_GU + l * 32768, 2 * NH, NH);
  }
  pack(iv1_w,           W16 + O_IV1A, NH, NH);
  pack(iv1_w + NH * NH, W16 + O_IV1B, NH, NH);
  pack(iv2_w, W16 + O_IV2, NH, NH);
  pack(cf1_w, W16 + O_CF1, 3 * NH, NH);
  pack(cf2_w, W16 + O_CF2, NH, NH);
  pack(ee1_w,           W16 + O_EE1A, NH, NH);
  pack(ee1_w + NH * NH, W16 + O_EE1B, NH, NH);
  pack(op_w,  W16 + O_OP, NN * NH, 64);
  cvt16<<<(NB * NH + 255) / 256, 256, 0, stream>>>(z_val, zv16, NB * NH);

  auto ggrid = [](int M, int N, int MT, int NT) {
    return ((M / (16 * MT)) * (N / (16 * NT)) + 3) / 4;
  };

  // ---- 1) input projection: x = [obs|act(+noise)] @ ip_w + ip_b  [2B,512] ----
  gemm_f16<2, 8, false, false><<<ggrid(2 * NB, NF, 2, 8), 128, 0, stream>>>(
      xcat16, NF + NA, W16 + O_IP, NF, ip_b, nullptr, nullptr,
      nullptr, x16, NF, 2 * NB, NF + NA);

  // ---- 2) h = gelu(ln(x @ ve1_w + ve1_b))  [2B,128] ----
  gemm_f16<2, 8, true, true><<<ggrid(2 * NB, NH, 2, 8), 128, 0, stream>>>(
      x16, NF, W16 + O_VE1, NH, ve1_b, ve_g, ve_be,
      nullptr, hbuf16, NH, 2 * NB, NF);

  // ---- 3) v = h @ ve2_w + ve2_b  [2B,1024] -> f32 + f16 ----
  gemm_f16<2, 8, false, false><<<ggrid(2 * NB, NN * NH, 2, 8), 128, 0, stream>>>(
      hbuf16, NH, W16 + O_VE2, NN * NH, ve2_b, nullptr, nullptr,
      vb32, v16, NN * NH, 2 * NB, NH);

  // ---- 4) edge scorer: hi/hj projections then fused pairwise adj ----
  gemm_f16<2, 8, false, false><<<ggrid(NB * NN, NH, 2, 8), 128, 0, stream>>>(
      v16, NH, W16 + O_ES1A, NH, nullptr, nullptr, nullptr,
      p1, nullptr, NH, NB * NN, NH);
  gemm_f16<2, 8, false, false><<<ggrid(NB * NN, NH, 2, 8), 128, 0, stream>>>(
      v16, NH, W16 + O_ES1B, NH, nullptr, nullptr, nullptr,
      p2, nullptr, NH, NB * NN, NH);
  pair_adj<<<NB, 256, 0, stream>>>(p1, p2, es1_b, es2_w, es2_b, temp, out + OFF_ADJ);

  // ---- 5) GNN message passing (3 layers), fused msg LN/GELU/agg ----
  for (int l = 0; l < NL; ++l) {
    const h16_t* vin = (l == 0) ? v16 : ((l == 1) ? w16a : w16b);
    h16_t* vout = (l == 1) ? w16b : w16a;
    float* vout32 = (l == NL - 1) ? (out + OFF_V) : nullptr;
    // hj = v @ gm_w[l,:H] ; hi = v @ gm_w[l,H:]
    gemm_f16<2, 8, false, false><<<ggrid(NB * NN, NH, 2, 8), 128, 0, stream>>>(
        vin, NH, W16 + O_GM + l * 32768, NH, nullptr, nullptr, nullptr,
        p1, nullptr, NH, NB * NN, NH);
    gemm_f16<2, 8, false, false><<<ggrid(NB * NN, NH, 2, 8), 128, 0, stream>>>(
        vin, NH, W16 + O_GM + l * 32768 + 16384, NH, nullptr, nullptr, nullptr,
        p2, nullptr, NH, NB * NN, NH);
    msg_agg<<<NB, 256, 0, stream>>>(p1, p2, gm_b + l * NH, gm_g + l * NH,
                                    gm_be + l * NH, out + OFF_ADJ, vin, acat16);
    // v = ln(cat([v,agg]) @ gu_w[l] + gu_b[l])
    gemm_f16<2, 8, true, false><<<ggrid(NB * NN, NH, 2, 8), 128, 0, stream>>>(
        acat16, 2 * NH, W16 + O_GU + l * 32768, NH, gu_b + l * NH,
        gu_g + l * NH, gu_be + l * NH, vout32, vout, NH, NB * NN, 2 * NH);
  }
  h16_t* vfin16 = w16a;
  float* vfin32 = out + OFF_V;

  // ---- 6) pairwise effect estimator ----
  gemm_f16<2, 8, false, false><<<ggrid(NB * NN, NH, 2, 8), 128, 0, stream>>>(
      vfin16, NH, W16 + O_EE1A, NH, nullptr, nullptr, nullptr,
      p1, nullptr, NH, NB * NN, NH);
  gemm_f16<2, 8, false, false><<<ggrid(NB * NN, NH, 2, 8), 128, 0, stream>>>(
      vfin16, NH, W16 + O_EE1B, NH, nullptr, nullptr, nullptr,
      p2, nullptr, NH, NB * NN, NH);
  pair_eff<<<NB, 256, 0, stream>>>(p1, p2, ee1_b, ee_g, ee_be, ee2_w, ee2_b,
                                   out + OFF_EFF);

  // ---- 7) intervention do(X_0 = z_val) ----
  h16_t* z16full = xcat16;  // xcat dead, reuse
  build_z<<<(int)(((size_t)NB * NN * NH + 255) / 256), 256, 0, stream>>>(
      vfin16, zv16, z16full);
  gemm_f16<2, 8, false, false><<<ggrid(NB, NH, 2, 8), 128, 0, stream>>>(
      zv16, NH, W16 + O_IV1A, NH, nullptr, nullptr, nullptr,
      hs32, nullptr, NH, NB, NH);                               // hs = z0 @ iv1_w[:H]
  gemm_f16<2, 8, false, false><<<ggrid(NB * NN, NH, 2, 8), 128, 0, stream>>>(
      z16full, NH, W16 + O_IV1B, NH, nullptr, nullptr, nullptr,
      p1, nullptr, NH, NB * NN, NH);                            // hz = z @ iv1_w[H:]
  h16_t* t16 = acat16;  // acat dead, reuse
  iv_pre<<<NB, 256, 0, stream>>>(hs32, p1, iv1_b, iv_g, iv_be, t16);
  gemm_f16<2, 8, false, false><<<ggrid(NB * NN, NH, 2, 8), 128, 0, stream>>>(
      t16, NH, W16 + O_IV2, NH, iv2_b, nullptr, nullptr,
      p2, nullptr, NH, NB * NN, NH);                            // delta
  iv_post<<<(int)(((size_t)NB * NN * NH + 255) / 256), 256, 0, stream>>>(
      p2, out + OFF_ADJ, vfin32, z_val, out + OFF_ITV);

  // ---- 8) counterfactual head ----
  const h16_t* vcf16 = v16 + (size_t)NB * NN * NH;   // cf rows of encode output
  const float* vcf32 = vb32 + (size_t)NB * NN * NH;
  build_cfin<<<(int)(((size_t)NB * 384 + 255) / 256), 256, 0, stream>>>(
      vfin16, vcf16, vfin32, vcf32, cfin16);
  gemm_f16<2, 8, true, true><<<ggrid(NB, NH, 2, 8), 128, 0, stream>>>(
      cfin16, 3 * NH, W16 + O_CF1, NH, cf1_b, cf_g, cf_be,
      nullptr, c116, NH, NB, 3 * NH);
  gemm_f16<2, 8, false, false><<<ggrid(NB, NH, 2, 8), 128, 0, stream>>>(
      c116, NH, W16 + O_CF2, NH, cf2_b, nullptr, nullptr,
      out + OFF_CF, nullptr, NH, NB, NH);

  // ---- 9) causal_repr = v.reshape(B,1024) @ op_w + op_b ----
  gemm_f16<2, 4, false, false><<<ggrid(NB, 64, 2, 4), 128, 0, stream>>>(
      vfin16, NN * NH, W16 + O_OP, 64, op_b, nullptr, nullptr,
      out + OFF_CR, nullptr, 64, NB, NN * NH);

  // ---- 10) adj.mean() ----
  reduce_partial<<<256, 256, 0, stream>>>(out + OFF_ADJ, part);
  reduce_final<<<1, 256, 0, stream>>>(part, out + OFF_MEAN);
}
